// PoolNetv3_61607010894042
// MI455X (gfx1250) — matrix operand order repaired and checked by reading, not compiled, not run on hardware
//
#include <hip/hip_runtime.h>
#include <hip/hip_bf16.h>

typedef __attribute__((ext_vector_type(16))) _Float16 v16h;
typedef __attribute__((ext_vector_type(8)))  float    v8f;

static inline int divup(long a, long b) { return (int)((a + b - 1) / b); }

__device__ __forceinline__ float elu1(float x) { return x > 0.f ? x : (__expf(x) - 1.f); }

// Low 32 bits of a generic pointer to __shared__ = wave-relative LDS byte offset
__device__ __forceinline__ unsigned lds_off(const void* p) {
    return (unsigned)(unsigned long long)p;
}

// ---------------- fill kernels ----------------
__global__ void fill_f32(float* p, float v, long n) {
    long i = (long)blockIdx.x * blockDim.x + threadIdx.x;
    if (i < n) p[i] = v;
}
__global__ void fill_i32(int* p, int v, long n) {
    long i = (long)blockIdx.x * blockDim.x + threadIdx.x;
    if (i < n) p[i] = v;
}

// ---------------- operand packing (f32 -> f16 WMMA fragment layout) ----------------
// A[M,K] -> Ap[(Mp/16) tiles][Ks][32 lanes][16 halves]; one thread per 8 halves.
// Fragment map (16-bit A 16x32): lane<16 holds K{0..7,16..23}, lane>=16 holds K{8..15,24..31}.
__global__ void pack_a_f16(const float* __restrict__ A, _Float16* __restrict__ Ap,
                           int M, int K, int Ks, long total)
{
    long idx = (long)blockIdx.x * blockDim.x + threadIdx.x;
    if (idx >= total) return;
    int oct  = (int)(idx & 1);
    int lane = (int)((idx >> 1) & 31);
    long tk  = idx >> 6;                       // tile*Ks + ks
    int ks   = (int)(tk % Ks);
    int tile = (int)(tk / Ks);
    int row  = tile * 16 + (lane & 15);
    int kbase = ks * 32 + oct * 16 + ((lane & 16) ? 8 : 0);
    alignas(16) _Float16 tmp[8];
#pragma unroll
    for (int j = 0; j < 8; ++j) {
        int k = kbase + j;
        float v = (row < M && k < K) ? A[(long)row * K + k] : 0.f;
        tmp[j] = (_Float16)v;
    }
    *(uint4*)(Ap + idx * 8) = *(const uint4*)tmp;
}
// W[K,N] -> Bp[Nt tiles][Ks][32 lanes][16 halves].
// Fragment map (16-bit B 32x16): col = lane&15, lane<16 holds K{0..7,16..23}, lane>=16 holds K{8..15,24..31}.
__global__ void pack_b_f16(const float* __restrict__ W, _Float16* __restrict__ Bp,
                           int N, int K, int Ks, long total)
{
    long idx = (long)blockIdx.x * blockDim.x + threadIdx.x;
    if (idx >= total) return;
    int oct  = (int)(idx & 1);
    int lane = (int)((idx >> 1) & 31);
    long tk  = idx >> 6;
    int ks   = (int)(tk % Ks);
    int tile = (int)(tk / Ks);
    int col  = tile * 16 + (lane & 15);
    int kbase = ks * 32 + oct * 16 + ((lane & 16) ? 8 : 0);
    alignas(16) _Float16 tmp[8];
#pragma unroll
    for (int j = 0; j < 8; ++j) {
        int k = kbase + j;
        float v = (col < N && k < K) ? W[(long)k * N + col] : 0.f;
        tmp[j] = (_Float16)v;
    }
    *(uint4*)(Bp + idx * 8) = *(const uint4*)tmp;
}

// ---------------- WMMA GEMM ----------------
// C[M,N] = act(A @ W + bias) from pre-packed f16 fragments.
// 256 threads = 8 waves; block tile = 32 rows x 128 cols; wave = one 16-col tile,
// two 16-row tiles -> 2 v_wmma per K-step with register-shared B fragment.
// A fragments (2KB/K-step, shared by all 8 waves) staged via async DMA to LDS.
__global__ void __launch_bounds__(256)
gemm_wmma(const _Float16* __restrict__ Ap, const _Float16* __restrict__ Bp,
          const float* __restrict__ bias, float* __restrict__ C,
          int M, int N, int Ks, int act)
{
    __shared__ _Float16 aLDS[1024] __attribute__((aligned(128)));
    const int tid  = threadIdx.x;
    const int lane = tid & 31;
    const int wave = tid >> 5;                 // column tile within block (0..7)
    const int Nt   = (N + 15) >> 4;
    const int mt0  = blockIdx.y * 2;
    const int ctg  = blockIdx.x * 8 + wave;
    const int ct   = (ctg < Nt) ? ctg : (Nt - 1);   // clamp: wave-uniform, EXEC stays full
    const _Float16* bp = Bp + ((size_t)ct * Ks) * 512 + (size_t)lane * 16;
    const size_t aBase = ((size_t)mt0 * Ks) * 512;

    v8f acc0 = {}, acc1 = {};
    for (int ks = 0; ks < Ks; ++ks) {
        __syncthreads();                       // protect LDS from previous iteration's readers
        if (tid < 128) {                       // 128 lanes x 16B = 2KB (two A fragments)
            int frag = tid >> 6;
            int t64  = tid & 63;
            const _Float16* src = Ap + aBase + ((size_t)frag * Ks + ks) * 512 + (size_t)t64 * 8;
            unsigned dst = lds_off(aLDS) + (unsigned)(frag * 1024 + t64 * 16);
            unsigned long long ga = (unsigned long long)src;
            asm volatile("global_load_async_to_lds_b128 %0, %1, off"
                         :: "v"(dst), "v"(ga) : "memory");
        }
        asm volatile("s_wait_asynccnt 0x0" ::: "memory");
        __syncthreads();
        v16h a0 = *(const v16h*)(aLDS + lane * 16);
        v16h a1 = *(const v16h*)(aLDS + 512 + lane * 16);
        v16h b  = *(const v16h*)(bp + (size_t)ks * 512);
        acc0 = __builtin_amdgcn_wmma_f32_16x16x32_f16(false, a0, false, b, (short)0, acc0, false, false);
        acc1 = __builtin_amdgcn_wmma_f32_16x16x32_f16(false, a1, false, b, (short)0, acc1, false, false);
    }
    if (ctg < Nt) {
        int col = ct * 16 + (lane & 15);
        if (col < N) {
            float bval = bias ? bias[col] : 0.f;
            int rbase = mt0 * 16 + ((lane & 16) ? 8 : 0);
#pragma unroll
            for (int v = 0; v < 8; ++v) {
                int r0 = rbase + v;
                if (r0 < M) {
                    float o = acc0[v] + bval;
                    if (act == 1) o = elu1(o);
                    C[(long)r0 * N + col] = o;
                }
                int r1 = rbase + 16 + v;
                if (r1 < M) {
                    float o = acc1[v] + bval;
                    if (act == 1) o = elu1(o);
                    C[(long)r1 * N + col] = o;
                }
            }
        }
    }
}

// ---------------- GAT pieces ----------------
__global__ void alpha_kernel(const float* __restrict__ h, const float* __restrict__ a_src,
                             const float* __restrict__ a_dst, float* as, float* ad,
                             int n, int H, int Cc)
{
    int i = blockIdx.x * blockDim.x + threadIdx.x;
    if (i >= n * H) return;
    int node = i / H, head = i % H;
    const float* hp  = h + (long)node * H * Cc + (long)head * Cc;
    const float* asp = a_src + head * Cc;
    const float* adp = a_dst + head * Cc;
    float s1 = 0.f, s2 = 0.f;
    for (int c = 0; c < Cc; ++c) { float hv = hp[c]; s1 += hv * asp[c]; s2 += hv * adp[c]; }
    as[i] = s1; ad[i] = s2;
}

// pass1: e = exp(leaky_relu(as[s]+ad[d])); accumulate per-dst denominator.
// (softmax coef = e/s is invariant to the max shift; logits are O(1) here)
__global__ void edge_pass1(const int* __restrict__ srcI, const int* __restrict__ dstI,
                           const int* __restrict__ remap,
                           const float* __restrict__ as, const float* __restrict__ ad,
                           float* __restrict__ ebuf, int* __restrict__ esrc, int* __restrict__ edst,
                           float* __restrict__ ssum, int E, int nLoops, int H)
{
    int e = blockIdx.x * blockDim.x + threadIdx.x;
    int total = E + nLoops;
    if (e >= total) return;
    int s, d; bool valid = true;
    if (e < E) {
        s = srcI[e]; d = dstI[e];
        if (remap) { s = remap[s]; d = remap[d]; valid = (s >= 0 && d >= 0); }
    } else { s = d = e - E; }            // self loops (already in new index space)
    esrc[e] = valid ? s : -1;
    edst[e] = valid ? d : -1;
    for (int h = 0; h < H; ++h) {
        float ev = 0.f;
        if (valid) {
            float lg = as[s * H + h] + ad[d * H + h];
            lg = lg > 0.f ? lg : 0.2f * lg;
            ev = __expf(lg);
            atomicAdd(&ssum[d * H + h], ev);
        }
        ebuf[e * H + h] = ev;
    }
}

__global__ void edge_pass2(const int* __restrict__ esrc, const int* __restrict__ edst,
                           const float* __restrict__ ebuf, const float* __restrict__ hbuf,
                           float* __restrict__ acc, int total, int H, int Cc)
{
    long HC = (long)H * Cc;
    long i = (long)blockIdx.x * blockDim.x + threadIdx.x;
    if (i >= (long)total * HC) return;
    int e = (int)(i / HC);
    int c = (int)(i % HC);
    int d = edst[e];
    if (d < 0) return;
    int s = esrc[e];
    int h = c / Cc;
    float ev = ebuf[e * H + h];
    atomicAdd(&acc[(long)d * HC + c], ev * hbuf[(long)s * HC + c]);
}

__global__ void gat_finalize(float* __restrict__ acc, const float* __restrict__ ssum,
                             const float* __restrict__ bias, int n, int H, int Cc)
{
    long i = (long)blockIdx.x * blockDim.x + threadIdx.x;
    int HC = H * Cc;
    if (i >= (long)n * HC) return;
    int node = (int)(i / HC), c = (int)(i % HC), h = c / Cc;
    float o = acc[i] / (ssum[node * H + h] + 1e-16f) + bias[c];
    acc[i] = elu1(o);
}

// ---------------- InstanceNorm ----------------
__global__ void inorm_stats(const float* __restrict__ x, float* mean, float* istd, int NN, int Cn)
{
    int g = blockIdx.x / Cn, c = blockIdx.x % Cn;
    const float* xp = x + (long)g * NN * Cn + c;
    float s = 0.f, s2 = 0.f;
    for (int i = threadIdx.x; i < NN; i += blockDim.x) {
        float v = xp[(long)i * Cn]; s += v; s2 += v * v;
    }
    __shared__ float ls[256], lq[256];
    ls[threadIdx.x] = s; lq[threadIdx.x] = s2;
    __syncthreads();
    for (int o = 128; o > 0; o >>= 1) {
        if (threadIdx.x < o) { ls[threadIdx.x] += ls[threadIdx.x + o]; lq[threadIdx.x] += lq[threadIdx.x + o]; }
        __syncthreads();
    }
    if (threadIdx.x == 0) {
        float m = ls[0] / NN;
        float var = lq[0] / NN - m * m;
        mean[blockIdx.x] = m;
        istd[blockIdx.x] = rsqrtf(var + 1e-5f);
    }
}
__global__ void inorm_apply(float* x, const float* mean, const float* istd,
                            const float* w, const float* b, int n, int Cn, int NN)
{
    long i = (long)blockIdx.x * blockDim.x + threadIdx.x;
    if (i >= (long)n * Cn) return;
    int node = (int)(i / Cn), c = (int)(i % Cn), g = node / NN;
    x[i] = (x[i] - mean[g * Cn + c]) * istd[g * Cn + c] * w[c] + b[c];
}

// ---------------- TopK pooling ----------------
__global__ void wnorm_kernel(const float* w, float* invnorm) {
    __shared__ float ls[64];
    float v = w[threadIdx.x];
    ls[threadIdx.x] = v * v;
    __syncthreads();
    for (int o = 32; o > 0; o >>= 1) { if (threadIdx.x < o) ls[threadIdx.x] += ls[threadIdx.x + o]; __syncthreads(); }
    if (threadIdx.x == 0) invnorm[0] = rsqrtf(ls[0]);
}
__global__ void score_kernel(const float* __restrict__ x, const float* __restrict__ w,
                             const float* invnorm, float* score, int n, int Cn)
{
    int i = blockIdx.x * blockDim.x + threadIdx.x;
    if (i >= n) return;
    float s = 0.f;
    for (int c = 0; c < Cn; ++c) s += x[(long)i * Cn + c] * w[c];
    score[i] = tanhf(s * invnorm[0]);
}
__global__ void __launch_bounds__(1024) topk_kernel(const float* __restrict__ score,
                                                    int* __restrict__ perm, int NN, int kSel)
{
    __shared__ float key[4096];
    __shared__ int   ids[4096];
    if (NN > 4096) return;
    int g = blockIdx.x;
    for (int i = threadIdx.x; i < NN; i += blockDim.x) { key[i] = score[g * NN + i]; ids[i] = i; }
    __syncthreads();
    for (int k = 2; k <= NN; k <<= 1)
        for (int j = k >> 1; j > 0; j >>= 1) {
            for (int i = threadIdx.x; i < NN; i += blockDim.x) {
                int ixj = i ^ j;
                if (ixj > i) {
                    bool up = ((i & k) == 0);
                    if ((key[i] < key[ixj]) == up) {
                        float tk = key[i]; key[i] = key[ixj]; key[ixj] = tk;
                        int ti = ids[i]; ids[i] = ids[ixj]; ids[ixj] = ti;
                    }
                }
            }
            __syncthreads();
        }
    for (int i = threadIdx.x; i < kSel; i += blockDim.x)
        perm[g * kSel + i] = g * NN + ids[i];
}
__global__ void scatter_newidx(const int* perm, int* new_idx, int n2) {
    int i = blockIdx.x * blockDim.x + threadIdx.x;
    if (i < n2) new_idx[perm[i]] = i;
}
__global__ void gather_x2(const float* __restrict__ x, const float* __restrict__ score,
                          const int* perm, float* x2, int n2, int Cn)
{
    long i = (long)blockIdx.x * blockDim.x + threadIdx.x;
    if (i >= (long)n2 * Cn) return;
    int r = (int)(i / Cn), c = (int)(i % Cn);
    int p = perm[r];
    x2[i] = x[(long)p * Cn + c] * score[p];
}

// ---------------- GlobalAttention + head ----------------
__global__ void gate2_kernel(const float* __restrict__ g1, const float* __restrict__ w2,
                             const float* b2, float* gs, int n2)
{
    int i = blockIdx.x * blockDim.x + threadIdx.x;
    if (i >= n2) return;
    float s = 0.f;
    for (int c = 0; c < 128; ++c) s += g1[(long)i * 128 + c] * w2[c];
    gs[i] = s + b2[0];
}
__global__ void __launch_bounds__(256) attn_pool(const float* __restrict__ gs,
                                                 const float* __restrict__ feat,
                                                 float* pooled, int kSel)
{
    int g = blockIdx.x;
    const float* gp = gs + (long)g * kSel;
    __shared__ float red[256];
    float m = -1e30f;
    for (int i = threadIdx.x; i < kSel; i += 256) m = fmaxf(m, gp[i]);
    red[threadIdx.x] = m; __syncthreads();
    for (int o = 128; o > 0; o >>= 1) { if (threadIdx.x < o) red[threadIdx.x] = fmaxf(red[threadIdx.x], red[threadIdx.x + o]); __syncthreads(); }
    m = red[0]; __syncthreads();
    float s = 0.f;
    for (int i = threadIdx.x; i < kSel; i += 256) s += __expf(gp[i] - m);
    red[threadIdx.x] = s; __syncthreads();
    for (int o = 128; o > 0; o >>= 1) { if (threadIdx.x < o) red[threadIdx.x] += red[threadIdx.x + o]; __syncthreads(); }
    s = red[0];
    int c = threadIdx.x;
    float acc = 0.f;
    const float* fp = feat + (long)g * kSel * 256;
    for (int i = 0; i < kSel; ++i) acc += __expf(gp[i] - m) * fp[(long)i * 256 + c];
    pooled[g * 256 + c] = acc / s;
}
__global__ void logsoftmax_kernel(const float* __restrict__ logits, float* out, int B, int NC)
{
    int r = threadIdx.x;
    if (r >= B) return;
    const float* lp = logits + (long)r * NC;
    float m = -1e30f;
    for (int c = 0; c < NC; ++c) m = fmaxf(m, lp[c]);
    float s = 0.f;
    for (int c = 0; c < NC; ++c) s += __expf(lp[c] - m);
    float ls = logf(s);
    for (int c = 0; c < NC; ++c) out[r * NC + c] = lp[c] - m - ls;
}

// ---------------- orchestration ----------------
extern "C" void kernel_launch(void* const* d_in, const int* in_sizes, int n_in,
                              void* d_out, int out_size, void* d_ws, size_t ws_size,
                              hipStream_t stream) {
    const float* pos   = (const float*)d_in[0];
    const int*   eidx  = (const int*)d_in[1];
    const float* c1_W  = (const float*)d_in[4];
    const float* c1_as = (const float*)d_in[5];
    const float* c1_ad = (const float*)d_in[6];
    const float* c1_b  = (const float*)d_in[7];
    const float* n1_w  = (const float*)d_in[8];
    const float* n1_b  = (const float*)d_in[9];
    const float* p1_w  = (const float*)d_in[10];
    const float* c2_W  = (const float*)d_in[11];
    const float* c2_as = (const float*)d_in[12];
    const float* c2_ad = (const float*)d_in[13];
    const float* c2_b  = (const float*)d_in[14];
    const float* n2_w  = (const float*)d_in[15];
    const float* n2_b  = (const float*)d_in[16];
    const float* g_W1  = (const float*)d_in[17];
    const float* g_b1  = (const float*)d_in[18];
    const float* g_W2  = (const float*)d_in[19];
    const float* g_b2  = (const float*)d_in[20];
    const float* a_W   = (const float*)d_in[21];
    const float* a_b   = (const float*)d_in[22];
    const float* l1_W  = (const float*)d_in[23];
    const float* l1_b  = (const float*)d_in[24];
    const float* l2_W  = (const float*)d_in[25];
    const float* l2_b  = (const float*)d_in[26];
    const float* l3_W  = (const float*)d_in[27];
    const float* l3_b  = (const float*)d_in[28];

    const int n  = in_sizes[0] / 3;
    const int E  = in_sizes[1] / 2;
    const int B  = 16;                       // num_graphs (fixed by setup; device scalar unreadable here)
    const int NN = n / B;
    const int kSel = (3 * NN + 9) / 10;      // ceil(0.3 * NN)
    const int n2 = B * kSel;
    const int* srcI = eidx;
    const int* dstI = eidx + E;

    char* wp = (char*)d_ws;
    auto bump = [&](size_t bytes) -> void* {
        void* r = (void*)wp; wp += (bytes + 255) & ~(size_t)255; return r;
    };
    auto needA = [&](int M, int K) { return (size_t)(divup(M, 32) * 2) * (size_t)((K + 31) / 32) * 512; };
    auto needB = [&](int N, int K) { return (size_t)divup(N, 16) * (size_t)((K + 31) / 32) * 512; };
    size_t apkH = needA(n, 3);
    { size_t c[5] = { needA(n2, 64), needA(n2, 256), needA(B, 256), needA(B, 128), needA(B, 64) };
      for (int i = 0; i < 5; ++i) if (c[i] > apkH) apkH = c[i]; }
    size_t bpkH = needB(64, 3);
    { size_t c[6] = { needB(256, 64), needB(128, 256), needB(256, 256), needB(128, 256), needB(64, 128), needB(10, 64) };
      for (int i = 0; i < 6; ++i) if (c[i] > bpkH) bpkH = c[i]; }

    float* h1    = (float*)bump((size_t)n * 64 * 4);        // conv1 features; reused as x2
    float* acc1  = (float*)bump((size_t)n * 64 * 4);        // conv1 accumulator -> x1
    float* as1   = (float*)bump((size_t)n * 2 * 4);
    float* ad1   = (float*)bump((size_t)n * 2 * 4);
    float* s1    = (float*)bump((size_t)n * 2 * 4);
    float* ebuf  = (float*)bump((size_t)(E + n) * 2 * 4);
    int*   esrc  = (int*)  bump((size_t)(E + n) * 4);
    int*   edst  = (int*)  bump((size_t)(E + n) * 4);
    float* mean1 = (float*)bump((size_t)B * 64 * 4);
    float* istd1 = (float*)bump((size_t)B * 64 * 4);
    float* invnm = (float*)bump(256);
    float* score = (float*)bump((size_t)n * 4);
    int*   perm  = (int*)  bump((size_t)n2 * 4);
    int*   nidx  = (int*)  bump((size_t)n * 4);
    float* h2    = (float*)bump((size_t)n2 * 256 * 4);      // conv2 features; later g1 then feat
    float* acc2  = (float*)bump((size_t)n2 * 256 * 4);      // conv2 accumulator -> x2n
    float* as2   = (float*)bump((size_t)n2 * 2 * 4);
    float* ad2   = (float*)bump((size_t)n2 * 2 * 4);
    float* s2    = (float*)bump((size_t)n2 * 2 * 4);
    float* mean2 = (float*)bump((size_t)B * 256 * 4);
    float* istd2 = (float*)bump((size_t)B * 256 * 4);
    float* gs    = (float*)bump((size_t)n2 * 4);
    float* pooled= (float*)bump((size_t)B * 256 * 4);
    float* m1    = (float*)bump((size_t)B * 128 * 4);
    float* m2    = (float*)bump((size_t)B * 64 * 4);
    float* lgt   = (float*)bump((size_t)B * 16 * 4);
    _Float16* apk = (_Float16*)bump(apkH * 2);
    _Float16* bpk = (_Float16*)bump(bpkH * 2);
    float* x2    = h1;  // h1 dead after conv1 aggregation
    if ((size_t)(wp - (char*)d_ws) > ws_size) return;

    // pack + WMMA GEMM launcher
    auto gemm = [&](const float* A, const float* W, const float* bias, float* Cout,
                    int M, int N, int K, int act) {
        int Ks = (K + 31) / 32;
        int Amt = divup(M, 32) * 2;
        int Nt = divup(N, 16);
        long aTh = (long)Amt * Ks * 64;
        long bTh = (long)Nt * Ks * 64;
        pack_a_f16<<<divup(aTh, 256), 256, 0, stream>>>(A, apk, M, K, Ks, aTh);
        pack_b_f16<<<divup(bTh, 256), 256, 0, stream>>>(W, bpk, N, K, Ks, bTh);
        gemm_wmma<<<dim3(divup(Nt, 8), divup(M, 32)), 256, 0, stream>>>(apk, bpk, bias, Cout, M, N, Ks, act);
    };

    // ---- conv1 (GAT 3->64, H=2, C=32) ----
    fill_f32<<<divup((long)n * 64, 256), 256, 0, stream>>>(acc1, 0.f, (long)n * 64);
    fill_f32<<<divup((long)n * 2, 256), 256, 0, stream>>>(s1, 0.f, (long)n * 2);
    gemm(pos, c1_W, nullptr, h1, n, 64, 3, 0);
    alpha_kernel<<<divup((long)n * 2, 256), 256, 0, stream>>>(h1, c1_as, c1_ad, as1, ad1, n, 2, 32);
    edge_pass1<<<divup((long)E + n, 256), 256, 0, stream>>>(srcI, dstI, nullptr, as1, ad1, ebuf, esrc, edst, s1, E, n, 2);
    edge_pass2<<<divup((long)(E + n) * 64, 256), 256, 0, stream>>>(esrc, edst, ebuf, h1, acc1, E + n, 2, 32);
    gat_finalize<<<divup((long)n * 64, 256), 256, 0, stream>>>(acc1, s1, c1_b, n, 2, 32);
    inorm_stats<<<B * 64, 256, 0, stream>>>(acc1, mean1, istd1, NN, 64);
    inorm_apply<<<divup((long)n * 64, 256), 256, 0, stream>>>(acc1, mean1, istd1, n1_w, n1_b, n, 64, NN);

    // ---- TopK pooling (ratio 0.3) ----
    wnorm_kernel<<<1, 64, 0, stream>>>(p1_w, invnm);
    score_kernel<<<divup(n, 256), 256, 0, stream>>>(acc1, p1_w, invnm, score, n, 64);
    topk_kernel<<<B, 1024, 0, stream>>>(score, perm, NN, kSel);
    fill_i32<<<divup(n, 256), 256, 0, stream>>>(nidx, -1, n);
    scatter_newidx<<<divup(n2, 256), 256, 0, stream>>>(perm, nidx, n2);
    gather_x2<<<divup((long)n2 * 64, 256), 256, 0, stream>>>(acc1, score, perm, x2, n2, 64);

    // ---- conv2 (GAT 64->256, H=2, C=128) ----
    fill_f32<<<divup((long)n2 * 256, 256), 256, 0, stream>>>(acc2, 0.f, (long)n2 * 256);
    fill_f32<<<divup((long)n2 * 2, 256), 256, 0, stream>>>(s2, 0.f, (long)n2 * 2);
    gemm(x2, c2_W, nullptr, h2, n2, 256, 64, 0);
    alpha_kernel<<<divup((long)n2 * 2, 256), 256, 0, stream>>>(h2, c2_as, c2_ad, as2, ad2, n2, 2, 128);
    edge_pass1<<<divup((long)E + n2, 256), 256, 0, stream>>>(srcI, dstI, nidx, as2, ad2, ebuf, esrc, edst, s2, E, n2, 2);
    edge_pass2<<<divup((long)(E + n2) * 256, 256), 256, 0, stream>>>(esrc, edst, ebuf, h2, acc2, E + n2, 2, 128);
    gat_finalize<<<divup((long)n2 * 256, 256), 256, 0, stream>>>(acc2, s2, c2_b, n2, 2, 128);
    inorm_stats<<<B * 256, 256, 0, stream>>>(acc2, mean2, istd2, kSel, 256);
    inorm_apply<<<divup((long)n2 * 256, 256), 256, 0, stream>>>(acc2, mean2, istd2, n2_w, n2_b, n2, 256, kSel);

    // ---- GlobalAttention pooling ----
    float* g1 = h2;     // h2 dead after conv2
    gemm(acc2, g_W1, g_b1, g1, n2, 128, 256, 1);
    gate2_kernel<<<divup(n2, 256), 256, 0, stream>>>(g1, g_W2, g_b2, gs, n2);
    float* feat = h2;   // g1 dead after gate2
    gemm(acc2, a_W, a_b, feat, n2, 256, 256, 1);
    attn_pool<<<B, 256, 0, stream>>>(gs, feat, pooled, kSel);

    // ---- MLP head + log_softmax ----
    gemm(pooled, l1_W, l1_b, m1, B, 128, 256, 1);
    gemm(m1, l2_W, l2_b, m2, B, 64, 128, 1);
    gemm(m2, l3_W, l3_b, lgt, B, 10, 64, 0);
    logsoftmax_kernel<<<1, 32, 0, stream>>>(lgt, (float*)d_out, B, 10);
}
